// nnue_15908558864663
// MI455X (gfx1250) — matrix operand order repaired and checked by reading, not compiled, not run on hardware
//
#include <hip/hip_runtime.h>

typedef __attribute__((ext_vector_type(16))) _Float16 v16h;
typedef __attribute__((ext_vector_type(8)))  _Float16 v8h;
typedef __attribute__((ext_vector_type(4)))  _Float16 v4h;
typedef __attribute__((ext_vector_type(8)))  float    v8f;
typedef __attribute__((ext_vector_type(4)))  float    v4f;

#define FT_IN  41600
#define K_HALF 256
#define FT_OUT 512
#define BATCH  8192
#define BUF_BYTES 16384   // 256 rows x 32 k x 2B, one k-step B slice

// ---------------------------------------------------------------------------
// Kernel 1: convert ft_w (f32, [256 x 41600]) -> f16 in workspace, same layout
// ---------------------------------------------------------------------------
__global__ __launch_bounds__(256) void cvt_weights(const float* __restrict__ w,
                                                   _Float16* __restrict__ o) {
  const int i = (blockIdx.x * 256 + threadIdx.x) * 4;   // total = 10,649,600, grid covers exactly
  v4f x = *(const v4f*)(w + i);
  v4h y;
  y[0] = (_Float16)x[0]; y[1] = (_Float16)x[1];
  y[2] = (_Float16)x[2]; y[3] = (_Float16)x[3];
  *(v4h*)(o + i) = y;
}

// ---------------------------------------------------------------------------
// CDNA5 async copy: per-lane 16B global -> LDS, tracked by ASYNCcnt
// ---------------------------------------------------------------------------
__device__ __forceinline__ void async_copy_b128(unsigned lds_off, const _Float16* g) {
  asm volatile("global_load_async_to_lds_b128 %0, %1, off"
               :
               : "v"(lds_off), "v"(g)
               : "memory");
}

__device__ __forceinline__ void wait_asynccnt0() {
  asm volatile("s_wait_asynccnt 0" ::: "memory");
}

// A fragment (16-bit A 16x32): lane holds row (lane&15);
// elems 0..7  = K = 8*half + {0..7}   (ap already includes 8*half)
// elems 8..15 = K = 16 + 8*half + {0..7}
__device__ __forceinline__ v16h load_a(const float* ap) {
  v4f a0 = *(const v4f*)(ap);
  v4f a1 = *(const v4f*)(ap + 4);
  v4f a2 = *(const v4f*)(ap + 16);
  v4f a3 = *(const v4f*)(ap + 20);
  v16h A;
#pragma unroll
  for (int i = 0; i < 4; ++i) {
    A[i]      = (_Float16)a0[i];
    A[4 + i]  = (_Float16)a1[i];
    A[8 + i]  = (_Float16)a2[i];
    A[12 + i] = (_Float16)a3[i];
  }
  return A;
}

// ---------------------------------------------------------------------------
// Kernel 2: feature-transform GEMM, WMMA f16 -> f32 accum, B staged in LDS via
// async-to-LDS double buffering.
//   grid = (BATCH/64, 1, 2), block = 128 (4 waves, 2M x 2N).
//   wave tile: M=32 (2 subtiles), N=128 (8 subtiles) -> 16 WMMAs / k-step.
// ---------------------------------------------------------------------------
__global__ __launch_bounds__(128) void ft_gemm(const float* __restrict__ feat1,
                                               const float* __restrict__ feat2,
                                               const _Float16* __restrict__ wf16,
                                               const float* __restrict__ ft_b,
                                               _Float16* __restrict__ ftout) {
  __shared__ _Float16 smem[2 * 256 * 32];   // two 16KB B slices

  const int z       = blockIdx.z;
  const float* feat = z ? feat2 : feat1;

  const int tid  = threadIdx.x;
  const int lane = tid & 31;
  const int wave = tid >> 5;           // 0..3
  const int wm   = wave >> 1;          // 0..1
  const int wn   = wave & 1;           // 0..1
  const int m0   = blockIdx.x * 64 + wm * 32;   // wave owns rows m0..m0+31
  const int n0   = wn * 128;                    // wave owns cols n0..n0+127
  const int half = lane >> 4;          // 0: lanes 0-15, 1: lanes 16-31
  const int lrow = lane & 15;

  // A stream (f32, converted in-register)
  const float* arow0 = feat + (size_t)(m0 + lrow) * FT_IN + (half << 3);
  const float* arow1 = arow0 + (size_t)16 * FT_IN;

  // B staging: each thread owns weight rows 2*tid and 2*tid+1 (64B per row/k-step)
  const _Float16* g0 = wf16 + (size_t)(2 * tid) * FT_IN;
  const _Float16* g1 = g0 + FT_IN;
  const unsigned  lbase = (unsigned)(size_t)(&smem[0]);   // low 32 bits = LDS offset
  const unsigned  my0   = lbase + (unsigned)(2 * tid) * 64;
  const unsigned  my1   = my0 + 64;

  // B fragment read index in LDS (elements): row n = n0+nt*16+lrow, K = 8*half+{0..7}, 16+8*half+{0..7}
  const int sread = (n0 + lrow) * 32 + (half << 3);

  v8f acc[2][8];
#pragma unroll
  for (int mi = 0; mi < 2; ++mi)
#pragma unroll
    for (int nt = 0; nt < 8; ++nt) acc[mi][nt] = (v8f)(0.0f);

  // Prologue: stage k=0 slice into buffer 0
#pragma unroll
  for (int c = 0; c < 4; ++c) {
    async_copy_b128(my0 + c * 16, g0 + c * 8);
    async_copy_b128(my1 + c * 16, g1 + c * 8);
  }

  int it = 0;
  for (int k = 0; k < FT_IN; k += 32, ++it) {
    const int buf = it & 1;

    // 1) my slice-copies for the current buffer are complete
    wait_asynccnt0();
    // 2) everyone's copies complete; everyone done reading the other buffer
    __syncthreads();
    // 3) kick off the next slice into the other buffer
    if (k + 32 < FT_IN) {
      const unsigned dst0 = my0 + (buf ^ 1) * BUF_BYTES;
      const unsigned dst1 = my1 + (buf ^ 1) * BUF_BYTES;
      const _Float16* s0 = g0 + k + 32;
      const _Float16* s1 = g1 + k + 32;
#pragma unroll
      for (int c = 0; c < 4; ++c) {
        async_copy_b128(dst0 + c * 16, s0 + c * 8);
        async_copy_b128(dst1 + c * 16, s1 + c * 8);
      }
    }

    // Prefetch the streaming feature rows ahead (into L2).
    __builtin_prefetch(arow0 + k + 256, 0, 0);
    __builtin_prefetch(arow1 + k + 256, 0, 0);

    // 4) A fragments from global (f32 -> f16)
    v16h A0 = load_a(arow0 + k);
    v16h A1 = load_a(arow1 + k);

    // 5) B fragments from LDS, then 16 back-to-back WMMAs
    const int sb = buf * (256 * 32) + sread;
    v16h Bv[8];
#pragma unroll
    for (int nt = 0; nt < 8; ++nt) {
      union { v16h v; v8h h[2]; } u;
      u.h[0] = *(const v8h*)&smem[sb + nt * 512];
      u.h[1] = *(const v8h*)&smem[sb + nt * 512 + 16];
      Bv[nt] = u.v;
    }
#pragma unroll
    for (int nt = 0; nt < 8; ++nt) {
      acc[0][nt] = __builtin_amdgcn_wmma_f32_16x16x32_f16(
          false, A0, false, Bv[nt], (short)0, acc[0][nt], false, false);
      acc[1][nt] = __builtin_amdgcn_wmma_f32_16x16x32_f16(
          false, A1, false, Bv[nt], (short)0, acc[1][nt], false, false);
    }
  }

  // Epilogue: C/D layout — VGPR j, lane L -> M = j + 8*(L>=16), N = L&15
#pragma unroll
  for (int mi = 0; mi < 2; ++mi) {
#pragma unroll
    for (int nt = 0; nt < 8; ++nt) {
      const int n   = n0 + nt * 16 + lrow;     // 0..255
      const float b = ft_b[n];
      const int col = z * K_HALF + n;          // 0..511
#pragma unroll
      for (int j = 0; j < 8; ++j) {
        const int row = m0 + mi * 16 + j + (half << 3);
        float v = acc[mi][nt][j] + b;
        v = fminf(fmaxf(v, 0.0f), 1.0f);
        ftout[(size_t)row * FT_OUT + col] = (_Float16)v;
      }
    }
  }
}

// ---------------------------------------------------------------------------
// Kernel 3: tail MLP 512 -> 32 -> 32 -> 1 (tiny; one thread per batch row)
// ---------------------------------------------------------------------------
__global__ __launch_bounds__(256) void tail(const _Float16* __restrict__ ftout,
                                            const float* __restrict__ h1_w,
                                            const float* __restrict__ h1_b,
                                            const float* __restrict__ h2_w,
                                            const float* __restrict__ h2_b,
                                            const float* __restrict__ out_w,
                                            const float* __restrict__ out_b,
                                            float* __restrict__ out) {
  const int row = blockIdx.x * 256 + threadIdx.x;
  const _Float16* f = ftout + (size_t)row * FT_OUT;

  float h1[32];
  for (int n = 0; n < 32; ++n) {
    float s = h1_b[n];
    const float* w = h1_w + n * FT_OUT;
#pragma unroll 8
    for (int k = 0; k < FT_OUT; ++k) s += (float)f[k] * w[k];
    h1[n] = fminf(fmaxf(s, 0.0f), 1.0f);
  }

  float h2[32];
  for (int n = 0; n < 32; ++n) {
    float s = h2_b[n];
#pragma unroll
    for (int k = 0; k < 32; ++k) s += h1[k] * h2_w[n * 32 + k];
    h2[n] = fminf(fmaxf(s, 0.0f), 1.0f);
  }

  float s = out_b[0];
#pragma unroll
  for (int k = 0; k < 32; ++k) s += h2[k] * out_w[k];
  out[row] = s;
}

// ---------------------------------------------------------------------------
extern "C" void kernel_launch(void* const* d_in, const int* in_sizes, int n_in,
                              void* d_out, int out_size, void* d_ws, size_t ws_size,
                              hipStream_t stream) {
  const float* features1 = (const float*)d_in[0];
  const float* features2 = (const float*)d_in[1];
  const float* ft_w      = (const float*)d_in[2];
  const float* ft_b      = (const float*)d_in[3];
  const float* h1_w      = (const float*)d_in[4];
  const float* h1_b      = (const float*)d_in[5];
  const float* h2_w      = (const float*)d_in[6];
  const float* h2_b      = (const float*)d_in[7];
  const float* out_w     = (const float*)d_in[8];
  const float* out_b     = (const float*)d_in[9];
  float* out             = (float*)d_out;

  _Float16* wf16  = (_Float16*)d_ws;                         // 256*41600 f16 = 21.3 MB
  _Float16* ftout = wf16 + (size_t)K_HALF * FT_IN;           // 8192*512  f16 =  8.0 MB

  // 1) ft_w f32 -> f16 (10,649,600 elems; 4 per thread; 10400 blocks exactly)
  cvt_weights<<<(K_HALF * FT_IN) / (256 * 4), 256, 0, stream>>>(ft_w, wf16);

  // 2) WMMA feature transform: grid.z picks features1/features2
  dim3 g(BATCH / 64, 1, 2);
  ft_gemm<<<g, 128, 0, stream>>>(features1, features2, wf16, ft_b, ftout);

  // 3) tail MLP
  tail<<<BATCH / 256, 256, 0, stream>>>(ftout, h1_w, h1_b, h2_w, h2_b,
                                        out_w, out_b, out);
}
